// SelfAttention_27023934226579
// MI455X (gfx1250) — compile-verified, matrix operand order checked by reading, not run on hardware
//
#include <hip/hip_runtime.h>

typedef __attribute__((ext_vector_type(16))) __bf16          v16bf;
typedef __attribute__((ext_vector_type(8)))  float           v8f;

#define S_LEN 4096
#define DDIM  512
#define LOG_GAMMA  (-0.040821994520255166f)   /* ln(0.96) */
#define INV_SQRT_D (0.044194173824159216f)    /* 1/sqrt(512) */
#define NEG_BIG    (-1e30f)

#if defined(__has_builtin)
#  if __has_builtin(__builtin_amdgcn_global_load_async_to_lds_b128)
#    define HAVE_ASYNC_LDS 1
#  endif
#endif

#if defined(HAVE_ASYNC_LDS)
typedef int nv4i __attribute__((vector_size(16)));            // native int vec4
typedef __attribute__((address_space(1))) nv4i* gv4i_p;       // global
typedef __attribute__((address_space(3))) nv4i* lv4i_p;       // LDS

__device__ __forceinline__ void async_copy16(const void* g, void* l) {
    __builtin_amdgcn_global_load_async_to_lds_b128((gv4i_p)g, (lv4i_p)l, 0, 0);
}
#endif

__device__ __forceinline__ void wait_async0() {
#if defined(HAVE_ASYNC_LDS)
#  if __has_builtin(__builtin_amdgcn_s_wait_asynccnt)
    __builtin_amdgcn_s_wait_asynccnt(0);
#  else
    asm volatile("s_wait_asynccnt 0x0" ::: "memory");
#  endif
#endif
}

// 32-byte WMMA operand: two contiguous 16-byte runs (ISA frag K-order
// per lane is halves [lhalf+0..7] and [lhalf+16..23] -> two b128 loads).
union V16 {
    uint4  q[2];
    v16bf  v;
};

// ---------------------------------------------------------------------------
// Kernel 0: one-off weight conversion.  Wq/Wk/Wv fp32 [k][n] -> bf16
// TRANSPOSED WT[m][n][k].  1.5 MB total, L2-resident for the QKV GEMM.
// grid = (512*512/256, 3), block = 256.  Coalesced reads over n.
// ---------------------------------------------------------------------------
__global__ __launch_bounds__(256) void wcvt_kernel(
    const float* __restrict__ Wq, const float* __restrict__ Wk,
    const float* __restrict__ Wv, __bf16* __restrict__ WT)
{
    const int m = blockIdx.y;
    const float* W = (m == 0) ? Wq : (m == 1) ? Wk : Wv;
    __bf16* dst = WT + (size_t)m * DDIM * DDIM;
    const int id = blockIdx.x * 256 + threadIdx.x;   // 0 .. 512*512-1
    const int k = id >> 9, n = id & 511;
    dst[(size_t)n * DDIM + k] = (__bf16)W[(size_t)k * DDIM + n];
}

// ---------------------------------------------------------------------------
// Kernel 1: fused QKV projection.  grid = (D/128, B*S/16), block = 256.
// One x tile (16x512 bf16) staged once in LDS and reused for ALL THREE
// matrices; B-fragments are two global_load_b128 from L2-resident WT.
// q/k stored row-major [b][seq][d]; V stored TRANSPOSED [b][d][seq] so
// each lane writes its 8 contiguous seq rows as one global_store_b128.
// ---------------------------------------------------------------------------
__global__ __launch_bounds__(256) void qkv_kernel(
    const float* __restrict__ x, const __bf16* __restrict__ WT,
    __bf16* __restrict__ qb, __bf16* __restrict__ kb, __bf16* __restrict__ vbT)
{
    __shared__ __align__(16) __bf16 sx[16 * DDIM];   // 16 KB, [row][k]

    const int tid  = threadIdx.x;
    const int lane = tid & 31;
    const int wave = tid >> 5;
    const int rowTile = blockIdx.y;                  // spans B*S/16
    const int colBase128 = blockIdx.x * 128;

    // Stage x tile: float4 global loads, packed bf16 stores.
    const float4* xrow = (const float4*)(x + (size_t)rowTile * 16 * DDIM);
    for (int i = tid; i < (16 * DDIM) / 4; i += 256) {
        float4 f = xrow[i];
        union { __bf16 h[4]; uint2 u; } pk;
        pk.h[0] = (__bf16)f.x; pk.h[1] = (__bf16)f.y;
        pk.h[2] = (__bf16)f.z; pk.h[3] = (__bf16)f.w;
        ((uint2*)sx)[i] = pk.u;
    }
    __syncthreads();

    const int lhalf = (lane >= 16) ? 8 : 0;
    const int arow  = lane & 15;
    const int ncol  = lane & 15;
    const int myCol = colBase128 + wave * 16 + ncol; // global output col
    v8f acc[3] = {{}, {}, {}};                       // q, k, v accumulators

    for (int ks = 0; ks < DDIM; ks += 32) {
        V16 a;
        const uint4* ap = (const uint4*)&sx[arow * DDIM + ks + lhalf];
        a.q[0] = ap[0];  a.q[1] = ap[2];             // halves +0..7, +16..23
        #pragma unroll
        for (int m = 0; m < 3; ++m) {
            V16 bfr;
            const uint4* bp = (const uint4*)(WT + (size_t)m * DDIM * DDIM +
                                             (size_t)myCol * DDIM + ks + lhalf);
            bfr.q[0] = bp[0];  bfr.q[1] = bp[2];
            acc[m] = __builtin_amdgcn_wmma_f32_16x16x32_bf16(
                         false, a.v, false, bfr.v, (short)0, acc[m], false, false);
        }
    }

    const int rbase = (lane >= 16) ? 8 : 0;
    // q and k: row-major [b][seq][d].
    #pragma unroll
    for (int r = 0; r < 8; ++r) {
        const size_t off = ((size_t)rowTile * 16 + rbase + r) * DDIM + myCol;
        qb[off] = (__bf16)acc[0][r];
        kb[off] = (__bf16)acc[1][r];
    }
    // v: transposed [b][d][seq]; 8 contiguous seq rows -> one b128 store.
    {
        union { __bf16 h[8]; uint4 q; } pk;
        #pragma unroll
        for (int r = 0; r < 8; ++r) pk.h[r] = (__bf16)acc[2][r];
        const int bIdx = (rowTile * 16) / S_LEN;
        const int seq0 = (rowTile * 16) % S_LEN + rbase;
        *(uint4*)&vbT[(size_t)bIdx * DDIM * S_LEN +
                      (size_t)myCol * S_LEN + seq0] = pk.q;
    }
}

// ---------------------------------------------------------------------------
// Kernel 2: flash attention with RetNet decay mask.
// grid = (S/16, B), block = 256 (8 waves). K chunk row-major in LDS,
// V chunk already transposed in global -> both staged with b128 copies
// (async-to-LDS via the gfx1250 builtin, tracked by ASYNCcnt).
// ---------------------------------------------------------------------------
__global__ __launch_bounds__(256) void attn_kernel(
    const __bf16* __restrict__ qb, const __bf16* __restrict__ kb,
    const __bf16* __restrict__ vbT, float* __restrict__ out)
{
    __shared__ __align__(16) __bf16 sK [32 * DDIM];  // 32 KB, [key][d]
    __shared__ __align__(16) __bf16 sVt[DDIM * 32];  // 32 KB, [d][key]
    __shared__ __align__(16) float  sS[16 * 32];     // fp32 score partials
    __shared__ __align__(16) __bf16 sP[16 * 32];     // probabilities
    __shared__ float sM[16], sL[16], sScale[16];

    const int tid   = threadIdx.x;
    const int lane  = tid & 31;
    const int wave  = tid >> 5;                      // 0..7
    const int qbase = blockIdx.x * 16;
    const int b     = blockIdx.y;
    const size_t baseOff = (size_t)b * S_LEN * DDIM;

    const int lhalf = (lane >= 16) ? 8 : 0;
    const int arow  = lane & 15;
    const int ncol  = lane & 15;
    const int rbase = (lane >= 16) ? 8 : 0;

    if (tid < 16) { sM[tid] = NEG_BIG; sL[tid] = 0.f; }

    // Persistent q A-fragments (this wave's 64-dim slice, 2 x K=32).
    v16bf qf[2];
    #pragma unroll
    for (int f = 0; f < 2; ++f) {
        V16 t;
        const uint4* qp = (const uint4*)(qb + baseOff +
                          (size_t)(qbase + arow) * DDIM + wave * 64 + f * 32 + lhalf);
        t.q[0] = qp[0];  t.q[1] = qp[2];
        qf[f] = t.v;
    }

    v8f acc[4] = {{}, {}, {}, {}};                   // cols: wave*64 + t*16 + ncol

    const int nChunks = (qbase + 47) / 32;           // causal frontier
    for (int c = 0; c < nChunks; ++c) {
        const int kb32 = c * 32;
        __syncthreads();                             // prev iter done with LDS

        // ---- Stage K [key][d] and V [d][key] chunks (b128 granularity) ----
        const __bf16* kSrc  = kb  + baseOff + (size_t)kb32 * DDIM;       // 2048 x 16B
        const __bf16* vSrcT = vbT + (size_t)b * DDIM * S_LEN + kb32;     // 512 rows x 4 x 16B
#if defined(HAVE_ASYNC_LDS)
        {
            for (int i = tid; i < 2048; i += 256)
                async_copy16((const char*)kSrc + (size_t)i * 16,
                             (char*)sK + (size_t)i * 16);
            for (int u = tid; u < 2048; u += 256) {
                const int d = u >> 2, p = u & 3;     // 4 x 16B per 32-key row
                async_copy16(vSrcT + (size_t)d * S_LEN + (size_t)p * 8,
                             sVt + d * 32 + p * 8);
            }
        }
#else
        {
            uint4* kDst = (uint4*)sK;
            for (int i = tid; i < 2048; i += 256)
                kDst[i] = ((const uint4*)kSrc)[i];
            for (int u = tid; u < 2048; u += 256) {
                const int d = u >> 2, p = u & 3;
                *(uint4*)&sVt[d * 32 + p * 8] =
                    *(const uint4*)&vSrcT[(size_t)d * S_LEN + (size_t)p * 8];
            }
            if (c + 1 < nChunks) {
                const char* nk = (const char*)(kSrc + 32 * DDIM);
                __builtin_prefetch(nk + tid * 128, 0, 1);
            }
        }
#endif
        for (int i = tid; i < 16 * 32; i += 256) sS[i] = 0.f;
        wait_async0();
        __syncthreads();

        // ---- Partial scores: this wave's 64 dims, two 16-key subtiles -----
        v8f s0 = {}, s1 = {};
        #pragma unroll
        for (int f = 0; f < 2; ++f) {
            const int d0 = wave * 64 + f * 32 + lhalf;
            V16 t0, t1;
            const uint4* p0 = (const uint4*)&sK[(size_t)ncol * DDIM + d0];
            const uint4* p1 = (const uint4*)&sK[(size_t)(16 + ncol) * DDIM + d0];
            t0.q[0] = p0[0];  t0.q[1] = p0[2];
            t1.q[0] = p1[0];  t1.q[1] = p1[2];
            s0 = __builtin_amdgcn_wmma_f32_16x16x32_bf16(
                     false, qf[f], false, t0.v, (short)0, s0, false, false);
            s1 = __builtin_amdgcn_wmma_f32_16x16x32_bf16(
                     false, qf[f], false, t1.v, (short)0, s1, false, false);
        }
        #pragma unroll
        for (int r = 0; r < 8; ++r) {
            atomicAdd(&sS[(rbase + r) * 32 + ncol],      s0[r]);
            atomicAdd(&sS[(rbase + r) * 32 + 16 + ncol], s1[r]);
        }
        __syncthreads();

        // ---- Online softmax + decay mask: wave 0, 32 lanes ----------------
        // lane = row + 16*half: each lane handles 16 keys of its row,
        // halves combined with __shfl_xor(16).
        if (wave == 0) {
            const int row = lane & 15;
            const int jh  = (lane >> 4) << 4;        // 0 or 16
            const int ig  = qbase + row;
            float sv[16];
            float mPart = NEG_BIG;
            #pragma unroll
            for (int j = 0; j < 16; ++j) {
                const int jg = kb32 + jh + j;
                float s = sS[row * 32 + jh + j] * INV_SQRT_D;
                s += (ig >= jg) ? (float)(ig - jg) * LOG_GAMMA : NEG_BIG;
                sv[j] = s;
                mPart = fmaxf(mPart, s);
            }
            const float mOld = sM[row];
            const float mNew = fmaxf(mOld, fmaxf(mPart, __shfl_xor(mPart, 16, 32)));
            float lPart = 0.f;
            #pragma unroll
            for (int j = 0; j < 16; ++j) {
                const float p = __expf(sv[j] - mNew);
                lPart += p;
                sP[row * 32 + jh + j] = (__bf16)p;
            }
            lPart += __shfl_xor(lPart, 16, 32);
            if (lane < 16) {
                const float scale = __expf(mOld - mNew);
                sL[row] = sL[row] * scale + lPart;
                sM[row] = mNew;
                sScale[row] = scale;
            }
        }
        __syncthreads();

        // ---- Rescale accumulators ----------------------------------------
        float sc[8];
        #pragma unroll
        for (int r = 0; r < 8; ++r) sc[r] = sScale[rbase + r];
        #pragma unroll
        for (int t = 0; t < 4; ++t)
            #pragma unroll
            for (int r = 0; r < 8; ++r) acc[t][r] *= sc[r];

        // ---- P (16x32) @ V (32 x 64-slice): 4 WMMAs -----------------------
        V16 pfu;
        const uint4* pp = (const uint4*)&sP[arow * 32 + lhalf];
        pfu.q[0] = pp[0];  pfu.q[1] = pp[2];
        const v16bf pf = pfu.v;
        #pragma unroll
        for (int t = 0; t < 4; ++t) {
            const int col = wave * 64 + t * 16 + ncol;
            V16 vt;
            const uint4* vp = (const uint4*)&sVt[col * 32 + lhalf];
            vt.q[0] = vp[0];  vt.q[1] = vp[2];
            acc[t] = __builtin_amdgcn_wmma_f32_16x16x32_bf16(
                         false, pf, false, vt.v, (short)0, acc[t], false, false);
        }
    }
    __syncthreads();

    // ---- Normalize and write fp32 output ---------------------------------
    float linv[8];
    #pragma unroll
    for (int r = 0; r < 8; ++r) linv[r] = 1.0f / sL[rbase + r];
    #pragma unroll
    for (int t = 0; t < 4; ++t) {
        const int col = wave * 64 + t * 16 + ncol;
        #pragma unroll
        for (int r = 0; r < 8; ++r)
            out[baseOff + (size_t)(qbase + rbase + r) * DDIM + col] = acc[t][r] * linv[r];
    }
}

// ---------------------------------------------------------------------------
extern "C" void kernel_launch(void* const* d_in, const int* in_sizes, int n_in,
                              void* d_out, int out_size, void* d_ws, size_t ws_size,
                              hipStream_t stream) {
    (void)in_sizes; (void)n_in; (void)out_size; (void)ws_size;
    const float* x  = (const float*)d_in[0];
    const float* Wq = (const float*)d_in[1];
    const float* Wk = (const float*)d_in[2];
    const float* Wv = (const float*)d_in[3];
    float* out = (float*)d_out;

    const int B = 4;
    const size_t nElem = (size_t)B * S_LEN * DDIM;
    __bf16* qb  = (__bf16*)d_ws;                     // bf16 q/k/v: ~48 MB
    __bf16* kb  = qb + nElem;
    __bf16* vbT = kb + nElem;                        // transposed [b][d][seq]
    __bf16* WT  = vbT + nElem;                       // 3 x 512 x 512 bf16 transposed

    dim3 g0((DDIM * DDIM) / 256, 3);
    wcvt_kernel<<<g0, 256, 0, stream>>>(Wq, Wk, Wv, WT);

    dim3 g1(DDIM / 128, (B * S_LEN) / 16);
    qkv_kernel<<<g1, 256, 0, stream>>>(x, WT, qb, kb, vbT);

    dim3 g2(S_LEN / 16, B);
    attn_kernel<<<g2, 256, 0, stream>>>(qb, kb, vbT, out);
}
